// DeepFM_37538014167469
// MI455X (gfx1250) — compile-verified, matrix operand order checked by reading, not run on hardware
//
#include <hip/hip_runtime.h>
#include <hip/hip_bf16.h>
#include <stdint.h>

// Problem constants
#define Bsz   4096
#define Fnum  39
#define Vocab 100000
#define Dd    16
#define Hh    400
#define K1    640   // F*D = 624 padded to 640 (20 k-tiles of 32)
#define K2    416   // H   = 400 padded to 416 (13 k-tiles of 32)
#define NPAD  448   // Wt row count padded (400 -> 448) so mainloop is unguarded

typedef __attribute__((ext_vector_type(16))) __bf16 v16bf;
typedef __attribute__((ext_vector_type(8)))  float  v8f;

__device__ __forceinline__ unsigned short f2bf(float f) {
  unsigned int u = __float_as_uint(f);
  u += 0x7FFFu + ((u >> 16) & 1u);   // round-to-nearest-even
  return (unsigned short)(u >> 16);
}

// ---------------------------------------------------------------------------
// Kernel 1: convert + transpose weights to bf16, zero-padded in K and N.
// Wt1: NPAD x K1 (row n = column n of W1), Wt2: NPAD x K2.
// ---------------------------------------------------------------------------
__global__ void convert_w_kernel(const float* __restrict__ W1,
                                 const float* __restrict__ W2,
                                 unsigned short* __restrict__ Wt1,
                                 unsigned short* __restrict__ Wt2)
{
  int i = blockIdx.x * blockDim.x + threadIdx.x;
  const int n1 = NPAD * K1;
  const int n2 = NPAD * K2;
  if (i < n1) {
    int n = i / K1, k = i % K1;
    Wt1[i] = (n < Hh && k < Fnum * Dd) ? f2bf(W1[k * Hh + n]) : (unsigned short)0;
  } else if (i < n1 + n2) {
    int j = i - n1;
    int n = j / K2, k = j % K2;
    Wt2[j] = (n < Hh && k < Hh) ? f2bf(W2[k * Hh + n]) : (unsigned short)0;
  }
}

// ---------------------------------------------------------------------------
// Kernel 2: embedding gather + first-order + FM second-order term.
// One wave32 per batch row: lanes = 2 features x 16 dims.
// ---------------------------------------------------------------------------
__global__ __launch_bounds__(256)
void gather_fm_kernel(const int*   __restrict__ Xi,
                      const float* __restrict__ Xv,
                      const float* __restrict__ fst_tables,
                      const float* __restrict__ sec_tables,
                      const float* __restrict__ bias,
                      unsigned short* __restrict__ A,
                      float* __restrict__ out)
{
  int wave = threadIdx.x >> 5;
  int lane = threadIdx.x & 31;
  int b    = blockIdx.x * 8 + wave;
  int sub  = lane >> 4;
  int d    = lane & 15;

  float s_acc = 0.f, ss_acc = 0.f, fst_acc = 0.f;

  for (int f0 = 0; f0 < Fnum; f0 += 2) {
    int f = f0 + sub;
    if (f < Fnum) {
      int   idx = Xi[b * Fnum + f];
      float val = Xv[b * Fnum + f];
      size_t row = (size_t)f * Vocab + (size_t)idx;
      float e = sec_tables[row * Dd + d] * val;     // coalesced 64B per feature
      s_acc  += e;
      ss_acc += e * e;
      A[(size_t)b * K1 + f * Dd + d] = f2bf(e);
      if (d == 0) fst_acc += fst_tables[row] * val;
    }
  }
  if (sub == 0) A[(size_t)b * K1 + Fnum * Dd + d] = 0;  // zero K-pad 624..639

  s_acc   += __shfl_xor(s_acc,   16, 32);
  ss_acc  += __shfl_xor(ss_acc,  16, 32);
  fst_acc += __shfl_xor(fst_acc, 16, 32);

  float fm = 0.5f * (s_acc * s_acc - ss_acc);
  #pragma unroll
  for (int off = 8; off >= 1; off >>= 1) fm += __shfl_xor(fm, off, 32);

  if (lane == 0) out[b] = fst_acc + fm + bias[0];
}

// ---------------------------------------------------------------------------
// Async staging helpers (CDNA5 ASYNCcnt path): global -> LDS DMA, 16B chunks.
// ---------------------------------------------------------------------------
__device__ __forceinline__ void async_b128(unsigned lds_addr, const void* gptr) {
  unsigned long long ga = (unsigned long long)(size_t)gptr;
  asm volatile("global_load_async_to_lds_b128 %0, %1, off"
               :: "v"(lds_addr), "v"(ga) : "memory");
}
__device__ __forceinline__ void wait_async3() {
  asm volatile("s_wait_asynccnt 3" ::: "memory");
}
__device__ __forceinline__ void wait_async0() {
  asm volatile("s_wait_asynccnt 0" ::: "memory");
}

// LDS layout (bf16 elements, row stride 40 to spread banks):
//   bufA[2] : 128 x 40  at elem offsets {0, 5120}
//   bufB[2] :  64 x 40  at elem offsets {10240, 12800}
#define LDS_ELEMS   15360
#define A_BUF_ELEMS 5120
#define B_BUF_BASE  10240
#define B_BUF_ELEMS 2560

// ---------------------------------------------------------------------------
// Kernels 3/4: tiled bf16 WMMA GEMM, C = A(MxK) * Bt(NxK)^T.
// Block tile 128x64, 8 waves, each wave 32x32 via 4x v_wmma_f32_16x16x32_bf16.
// Double-buffered LDS fed by global_load_async_to_lds_b128 (3 per thread/tile).
// MODE 0: Hout[m][n] = bf16(relu(acc + bias[n])), zero-fill n in [Nrows, ldh)
// MODE 1: out[m] += sum_n relu(acc + bias[n])   (float atomics)
// ---------------------------------------------------------------------------
template <int MODE>
__global__ __launch_bounds__(256)
void gemm_kernel(const unsigned short* __restrict__ A, int Kpad,
                 const unsigned short* __restrict__ Bt, int Nrows,
                 const float* __restrict__ bias,
                 unsigned short* __restrict__ Hout, int ldh,
                 float* __restrict__ out)
{
  __shared__ __align__(16) unsigned short smem[LDS_ELEMS];

  const int tid  = threadIdx.x;
  const int wave = tid >> 5, lane = tid & 31;
  const int wm   = wave >> 1;          // 0..3 : 32-row M sub-tile
  const int wn   = wave & 1;           // 0..1 : 32-col N sub-tile
  const int sub  = lane >> 4;          // K-half selector (ISA bf16 layout)
  const int l16  = lane & 15;

  const int m0 = blockIdx.x * 128;
  const int n0 = blockIdx.y * 64;
  const int nk = Kpad >> 5;

  // Low 32 bits of the generic pointer = LDS byte offset (HW adds LDS_BASE).
  const unsigned ldsBase = (unsigned)(size_t)(&smem[0]);

  // Per-thread staging chunks: A tile 128x32 = 512 x16B (2/thread),
  //                            B tile  64x32 = 256 x16B (1/thread).
  const int arow0 = tid >> 2;            // 0..63
  const int arow1 = arow0 + 64;          // 64..127
  const int acol  = (tid & 3) * 8;       // 0,8,16,24
  const int brow  = tid >> 2;
  const int bcol  = acol;

  auto prefetch = [&](int t, int bi) {
    int k0 = t * 32;
    async_b128(ldsBase + (unsigned)((bi * A_BUF_ELEMS + arow0 * 40 + acol) * 2),
               A + (size_t)(m0 + arow0) * Kpad + k0 + acol);
    async_b128(ldsBase + (unsigned)((bi * A_BUF_ELEMS + arow1 * 40 + acol) * 2),
               A + (size_t)(m0 + arow1) * Kpad + k0 + acol);
    async_b128(ldsBase + (unsigned)((B_BUF_BASE + bi * B_BUF_ELEMS + brow * 40 + bcol) * 2),
               Bt + (size_t)(n0 + brow) * Kpad + k0 + bcol);
  };

  v8f acc00 = {}, acc01 = {}, acc10 = {}, acc11 = {};

  prefetch(0, 0);
  prefetch(1, 1);

  for (int i = 0; i < nk; ++i) {
    if (i + 1 < nk) wait_async3();   // tile i complete (in-order done)
    else            wait_async0();
    __syncthreads();                 // tile i visible to all waves

    const unsigned short* bufA = smem + (i & 1) * A_BUF_ELEMS;
    const unsigned short* bufB = smem + B_BUF_BASE + (i & 1) * B_BUF_ELEMS;

    // ISA 16-bit fragment layout: lane<16 -> K{0..7,16..23}, lane>=16 -> K{8..15,24..31}
    union Frag { uint4 q[2]; v16bf v; };
    Frag a0, a1, b0, b1;
    int ar0 = wm * 32 + l16, ar1 = ar0 + 16;
    a0.q[0] = *(const uint4*)(bufA + ar0 * 40 + sub * 8);
    a0.q[1] = *(const uint4*)(bufA + ar0 * 40 + 16 + sub * 8);
    a1.q[0] = *(const uint4*)(bufA + ar1 * 40 + sub * 8);
    a1.q[1] = *(const uint4*)(bufA + ar1 * 40 + 16 + sub * 8);
    int br0 = wn * 32 + l16, br1 = br0 + 16;
    b0.q[0] = *(const uint4*)(bufB + br0 * 40 + sub * 8);
    b0.q[1] = *(const uint4*)(bufB + br0 * 40 + 16 + sub * 8);
    b1.q[0] = *(const uint4*)(bufB + br1 * 40 + sub * 8);
    b1.q[1] = *(const uint4*)(bufB + br1 * 40 + 16 + sub * 8);

    acc00 = __builtin_amdgcn_wmma_f32_16x16x32_bf16(false, a0.v, false, b0.v,
                                                    (short)0, acc00, false, false);
    acc01 = __builtin_amdgcn_wmma_f32_16x16x32_bf16(false, a0.v, false, b1.v,
                                                    (short)0, acc01, false, false);
    acc10 = __builtin_amdgcn_wmma_f32_16x16x32_bf16(false, a1.v, false, b0.v,
                                                    (short)0, acc10, false, false);
    acc11 = __builtin_amdgcn_wmma_f32_16x16x32_bf16(false, a1.v, false, b1.v,
                                                    (short)0, acc11, false, false);

    __syncthreads();                 // all waves done reading buffer (i&1)
    if (i + 2 < nk) prefetch(i + 2, i & 1);
  }

  // C layout: VGPR r, lanes 0-15 -> row r, lanes 16-31 -> row 8+r; n = lane%16.
  const int mb0 = m0 + wm * 32 + (sub << 3);   // rows of acc00/acc01 (+r)
  const int mb1 = mb0 + 16;                    // rows of acc10/acc11 (+r)
  const int nc0 = n0 + wn * 32 + l16;
  const int nc1 = nc0 + 16;
  const float bv0 = (nc0 < Nrows) ? bias[nc0] : 0.f;
  const float bv1 = (nc1 < Nrows) ? bias[nc1] : 0.f;

  if (MODE == 0) {
    auto store_tile = [&](const v8f& acc, int mb, int nc, float bv) {
      #pragma unroll
      for (int r = 0; r < 8; ++r) {
        float v = fmaxf(acc[r] + bv, 0.f);
        int m = mb + r;
        if (nc < Nrows)    Hout[(size_t)m * ldh + nc] = f2bf(v);
        else if (nc < ldh) Hout[(size_t)m * ldh + nc] = 0;
      }
    };
    store_tile(acc00, mb0, nc0, bv0);
    store_tile(acc01, mb0, nc1, bv1);
    store_tile(acc10, mb1, nc0, bv0);
    store_tile(acc11, mb1, nc1, bv1);
  } else {
    float rsA[8], rsB[8];
    #pragma unroll
    for (int r = 0; r < 8; ++r) {
      float v00 = (nc0 < Nrows) ? fmaxf(acc00[r] + bv0, 0.f) : 0.f;
      float v01 = (nc1 < Nrows) ? fmaxf(acc01[r] + bv1, 0.f) : 0.f;
      float v10 = (nc0 < Nrows) ? fmaxf(acc10[r] + bv0, 0.f) : 0.f;
      float v11 = (nc1 < Nrows) ? fmaxf(acc11[r] + bv1, 0.f) : 0.f;
      rsA[r] = v00 + v01;
      rsB[r] = v10 + v11;
    }
    #pragma unroll
    for (int off = 8; off >= 1; off >>= 1) {
      #pragma unroll
      for (int r = 0; r < 8; ++r) {
        rsA[r] += __shfl_xor(rsA[r], off, 32);
        rsB[r] += __shfl_xor(rsB[r], off, 32);
      }
    }
    if (l16 == 0) {
      #pragma unroll
      for (int r = 0; r < 8; ++r) {
        atomicAdd(&out[mb0 + r], rsA[r]);
        atomicAdd(&out[mb1 + r], rsB[r]);
      }
    }
  }
}

// ---------------------------------------------------------------------------
extern "C" void kernel_launch(void* const* d_in, const int* in_sizes, int n_in,
                              void* d_out, int out_size, void* d_ws, size_t ws_size,
                              hipStream_t stream)
{
  (void)in_sizes; (void)n_in; (void)out_size; (void)ws_size;

  const int*   Xi   = (const int*)  d_in[0];
  const float* Xv   = (const float*)d_in[1];
  const float* fstT = (const float*)d_in[2];
  const float* secT = (const float*)d_in[3];
  const float* W1   = (const float*)d_in[4];
  const float* b1   = (const float*)d_in[5];
  const float* W2   = (const float*)d_in[6];
  const float* b2   = (const float*)d_in[7];
  const float* bias = (const float*)d_in[8];
  float* out = (float*)d_out;

  char* ws = (char*)d_ws;
  unsigned short* Abuf = (unsigned short*)ws;                          // 4096*640*2
  unsigned short* Hbuf = (unsigned short*)(ws + (size_t)Bsz * K1 * 2); // 4096*416*2
  unsigned short* Wt1  = (unsigned short*)(ws + (size_t)Bsz * K1 * 2
                                              + (size_t)Bsz * K2 * 2); // 448*640*2
  unsigned short* Wt2  = Wt1 + (size_t)NPAD * K1;                      // 448*416*2

  int convN = NPAD * K1 + NPAD * K2;
  convert_w_kernel<<<(convN + 255) / 256, 256, 0, stream>>>(W1, W2, Wt1, Wt2);

  gather_fm_kernel<<<Bsz / 8, 256, 0, stream>>>(Xi, Xv, fstT, secT, bias, Abuf, out);

  // GEMM1: h = relu(sec_flat @ W1 + b1)   (4096 x 400, stored padded to 416)
  gemm_kernel<0><<<dim3(Bsz / 128, 7), 256, 0, stream>>>(
      Abuf, K1, Wt1, Hh, b1, Hbuf, K2, nullptr);

  // GEMM2: out[b] += sum(relu(h @ W2 + b2))
  gemm_kernel<1><<<dim3(Bsz / 128, 7), 256, 0, stream>>>(
      Hbuf, K2, Wt2, Hh, b2, nullptr, 0, out);
}
